// NeuralODEEventSLAM_3427383902462
// MI455X (gfx1250) — matrix operand order repaired and checked by reading, not compiled, not run on hardware
//
#include <hip/hip_runtime.h>
#include <hip/hip_bf16.h>

// ---------------------------------------------------------------------------
// NeuralODE Event SLAM forward pass for gfx1250 (MI455X).
// All large GEMMs run on v_wmma_f32_16x16x32_f16 (wave32 WMMA).
//
// d_in layout assumption (top-level dict insertion order, params flattened as
// a jax pytree: dict keys sorted, {"b","w"} -> b first):
//  0 events_left [16,64,128,4] f32    1 events_right
//  2 counts_left [16,64] i32          3 counts_right
//  4 agg0.b  5 agg0.w(64x64)  6 agg1.b  7 agg1.w(64x64)
//  8 attn.k.b  9 attn.k.w 10 attn.o.b 11 attn.o.w 12 attn.q.b 13 attn.q.w
// 14 attn.v.b 15 attn.v.w
// 16 e0.b 17 e0.w(128x4) 18 e1.b 19 e1.w(128x128) 20 e2.b 21 e2.w(64x128)
// 22 gate.b(256) 23 gate.w(256x64)
// 24 u0.b 25 u0.w(256x320) 26 u1.b 27 u1.w(256x256)
// 28 fuse_w(64x128) 29 h0(256)
// 30 d0.b 31 d0.w(256x288) 32 d1.b 33 d1.w(256x256) 34 d2.b 35 d2.w(256x256)
// 36 ot0.b(32) 37 ot0.w(32x1) 38 ot1.b(32) 39 ot1.w(32x32)
// 40 rot0.b 41 rot0.w(128x256) 42 rot1.b(4) 43 rot1.w(4x128)
// 44 tr0.b 45 tr0.w(128x256) 46 tr1.b(3) 47 tr1.w(3x128)
// d_out: poses [16,64,7] f32 followed by hs [16,64,256] f32.
// ---------------------------------------------------------------------------

#define NB 16
#define NS 64
#define NE 128
#define NHID 256
#define NEV 64
#define NHEADS 4

typedef __attribute__((ext_vector_type(16))) _Float16 v16h;
typedef __attribute__((ext_vector_type(8)))  float    v8f;

#define DEVI static __device__ __forceinline__

DEVI v8f wmma_f16(v16h a, v16h b, v8f c) {
  return __builtin_amdgcn_wmma_f32_16x16x32_f16(false, a, false, b, (short)0, c,
                                                false, false);
}

// A fragment: 16x32 f16, source row-major [m][k], row stride in elements.
// ISA layout: lane half (l>>4) offsets k by 8; VGPR v: k=(v>>2)*16+half*8+(v&3)*2.
DEVI v16h frag_a_h(const _Float16* p, int stride, int lane, int kmax) {
  v16h r; int m = lane & 15, hl = lane >> 4;
#pragma unroll
  for (int v = 0; v < 8; ++v) {
    int k = ((v >> 2) << 4) + (hl << 3) + ((v & 3) << 1);
    r[2*v]   = (k     < kmax) ? p[m*stride + k]     : (_Float16)0.f;
    r[2*v+1] = (k + 1 < kmax) ? p[m*stride + k + 1] : (_Float16)0.f;
  }
  return r;
}

// A fragment from f32 source (converted on the fly).
DEVI v16h frag_a_f(const float* p, int stride, int lane, int kmax) {
  v16h r; int m = lane & 15, hl = lane >> 4;
#pragma unroll
  for (int v = 0; v < 8; ++v) {
    int k = ((v >> 2) << 4) + (hl << 3) + ((v & 3) << 1);
    r[2*v]   = (k     < kmax) ? (_Float16)p[m*stride + k]     : (_Float16)0.f;
    r[2*v+1] = (k + 1 < kmax) ? (_Float16)p[m*stride + k + 1] : (_Float16)0.f;
  }
  return r;
}

// B fragment (32x16), source laid out [n][k] (weight matrices [out][in]).
// B layout matches A: VGPR v: k=(v>>2)*16+half*8+(v&3)*2.
DEVI v16h frag_b_nk(const _Float16* p, int stride, int lane, int kmax) {
  v16h r; int n = lane & 15, hl = lane >> 4;
#pragma unroll
  for (int v = 0; v < 8; ++v) {
    int k = ((v >> 2) << 4) + (hl << 3) + ((v & 3) << 1);
    r[2*v]   = (k     < kmax) ? p[n*stride + k]     : (_Float16)0.f;
    r[2*v+1] = (k + 1 < kmax) ? p[n*stride + k + 1] : (_Float16)0.f;
  }
  return r;
}

// B fragment, source laid out [k][n] (e.g. V matrix rows = K dimension).
DEVI v16h frag_b_kn(const _Float16* p, int stride, int lane, int kmax) {
  v16h r; int n = lane & 15, hl = lane >> 4;
#pragma unroll
  for (int v = 0; v < 8; ++v) {
    int k = ((v >> 2) << 4) + (hl << 3) + ((v & 3) << 1);
    r[2*v]   = (k     < kmax) ? p[k*stride + n]       : (_Float16)0.f;
    r[2*v+1] = (k + 1 < kmax) ? p[(k + 1)*stride + n] : (_Float16)0.f;
  }
  return r;
}

// --------------------------- weight f16 conversion --------------------------
__global__ void cvt_pad_kernel(const float* __restrict__ src,
                               _Float16* __restrict__ dst,
                               int rows, int cols, int pcols) {
  int i = blockIdx.x * 256 + threadIdx.x;
  if (i >= rows * pcols) return;
  int r = i / pcols, c = i % pcols;
  dst[i] = (c < cols) ? (_Float16)src[r*cols + c] : (_Float16)0.f;
}

// ------------------------------ event embedding ------------------------------
// rows = 2*B*S*E = 262144; 16-row tile per wave, 4 waves per workgroup.
__global__ void embed_kernel(const float* __restrict__ evL,
                             const float* __restrict__ evR,
                             const _Float16* __restrict__ w0, const float* __restrict__ b0,
                             const _Float16* __restrict__ w1, const float* __restrict__ b1,
                             const _Float16* __restrict__ w2, const float* __restrict__ b2,
                             _Float16* __restrict__ fout) {
  __shared__ _Float16 h1[4][16][128];
  __shared__ _Float16 h2[4][16][128];
  int wave = threadIdx.x >> 5, lane = threadIdx.x & 31;
  int nloc = lane & 15, hl = lane >> 4;
  int tile = blockIdx.x * 4 + wave;
  int rowbase = tile * 16;
  const int perSide = NB * NS * NE;
  const float* ev = (rowbase < perSide) ? (evL + rowbase * 4)
                                        : (evR + (rowbase - perSide) * 4);
  v16h a0 = frag_a_f(ev, 4, lane, 4);       // K=4 padded to 32 with zeros

  // layer 1: 4 -> 128, relu
#pragma unroll
  for (int nt = 0; nt < 8; ++nt) {
    v8f acc = {};
    acc = wmma_f16(a0, frag_b_nk(w0 + nt*16*32, 32, lane, 32), acc);
    int n = nt*16 + nloc; float bias = b0[n];
#pragma unroll
    for (int r = 0; r < 8; ++r)
      h1[wave][hl*8 + r][n] = (_Float16)fmaxf(acc[r] + bias, 0.f);
  }
  __syncthreads();
  // layer 2: 128 -> 128, relu
#pragma unroll
  for (int nt = 0; nt < 8; ++nt) {
    v8f acc = {};
#pragma unroll
    for (int kt = 0; kt < 4; ++kt)
      acc = wmma_f16(frag_a_h(&h1[wave][0][kt*32], 128, lane, 32),
                     frag_b_nk(w1 + nt*16*128 + kt*32, 128, lane, 32), acc);
    int n = nt*16 + nloc; float bias = b1[n];
#pragma unroll
    for (int r = 0; r < 8; ++r)
      h2[wave][hl*8 + r][n] = (_Float16)fmaxf(acc[r] + bias, 0.f);
  }
  __syncthreads();
  // layer 3: 128 -> 64
#pragma unroll
  for (int nt = 0; nt < 4; ++nt) {
    v8f acc = {};
#pragma unroll
    for (int kt = 0; kt < 4; ++kt)
      acc = wmma_f16(frag_a_h(&h2[wave][0][kt*32], 128, lane, 32),
                     frag_b_nk(w2 + nt*16*128 + kt*32, 128, lane, 32), acc);
    int n = nt*16 + nloc; float bias = b2[n];
#pragma unroll
    for (int r = 0; r < 8; ++r)
      fout[(rowbase + hl*8 + r) * 64 + n] = (_Float16)(acc[r] + bias);
  }
}

// ------------------------------ self attention -------------------------------
// One workgroup (8 waves) per (side,b,s) slice. LDS kept < 64KB:
// K/V f16 (32KB) + 32-row Q block (4KB) + 32x128 f32 score panel (16KB).
__global__ void attn_kernel(const _Float16* __restrict__ f,
                            const _Float16* __restrict__ wq, const float* __restrict__ bq,
                            const _Float16* __restrict__ wk, const float* __restrict__ bk,
                            const _Float16* __restrict__ wv, const float* __restrict__ bv,
                            const int* __restrict__ cntL, const int* __restrict__ cntR,
                            _Float16* __restrict__ oout) {
  __shared__ _Float16 kv[2][128][64];
  __shared__ _Float16 qb[32][64];
  __shared__ float    sc[32][128];
  int slice = blockIdx.x;
  int wave = threadIdx.x >> 5, lane = threadIdx.x & 31;
  int nloc = lane & 15, hl = lane >> 4;
  int cnt = (slice < NB*NS) ? cntL[slice] : cntR[slice - NB*NS];
  const _Float16* fs = f + slice * NE * 64;
  _Float16* os = oout + slice * NE * 64;

  { // K and V projections: wave -> row tile `wave`
    v16h A0 = frag_a_h(fs + wave*16*64,      64, lane, 32);
    v16h A1 = frag_a_h(fs + wave*16*64 + 32, 64, lane, 32);
    const _Float16* wm[2] = { wk, wv };
    const float*    bm[2] = { bk, bv };
#pragma unroll
    for (int p = 0; p < 2; ++p)
#pragma unroll
      for (int nt = 0; nt < 4; ++nt) {
        v8f acc = {};
        acc = wmma_f16(A0, frag_b_nk(wm[p] + nt*16*64,      64, lane, 32), acc);
        acc = wmma_f16(A1, frag_b_nk(wm[p] + nt*16*64 + 32, 64, lane, 32), acc);
        int n = nt*16 + nloc; float bias = bm[p][n];
#pragma unroll
        for (int r = 0; r < 8; ++r)
          kv[p][wave*16 + hl*8 + r][n] = (_Float16)(acc[r] + bias);
      }
  }
  __syncthreads();

  for (int qblk = 0; qblk < 4; ++qblk) {
    { // Q block (32 rows): 2 row tiles x 4 n-tiles over 8 waves
      int rt = wave >> 2, nt = wave & 3;
      int grow = qblk*32 + rt*16;
      v8f acc = {};
      acc = wmma_f16(frag_a_h(fs + grow*64,      64, lane, 32),
                     frag_b_nk(wq + nt*16*64,      64, lane, 32), acc);
      acc = wmma_f16(frag_a_h(fs + grow*64 + 32, 64, lane, 32),
                     frag_b_nk(wq + nt*16*64 + 32, 64, lane, 32), acc);
      int n = nt*16 + nloc; float bias = bq[n];
#pragma unroll
      for (int r = 0; r < 8; ++r)
        qb[rt*16 + hl*8 + r][n] = (_Float16)(acc[r] + bias);
    }
    __syncthreads();
    for (int h = 0; h < NHEADS; ++h) {
      // scores: 2 q-tiles x 8 k-tiles = 16 jobs, 2 per wave (K=16 zero-padded)
#pragma unroll
      for (int i = 0; i < 2; ++i) {
        int job = wave*2 + i;
        int qt = job >> 3, kt = job & 7;
        v8f acc = {};
        acc = wmma_f16(frag_a_h(&qb[qt*16][h*16], 64, lane, 16),
                       frag_b_nk(&kv[0][kt*16][h*16], 64, lane, 16), acc);
        int kcol = kt*16 + nloc;
#pragma unroll
        for (int r = 0; r < 8; ++r)
          sc[qt*16 + hl*8 + r][kcol] = (kcol < cnt) ? acc[r]*0.25f : -1e9f;
      }
      __syncthreads();
      if (threadIdx.x < 32) {  // row-wise masked softmax
        float* row = sc[threadIdx.x];
        float mx = -3.0e38f;
        for (int j = 0; j < 128; ++j) mx = fmaxf(mx, row[j]);
        float sum = 0.f;
        for (int j = 0; j < 128; ++j) { float e = __expf(row[j]-mx); row[j] = e; sum += e; }
        float inv = 1.f / sum;
        for (int j = 0; j < 128; ++j) row[j] *= inv;
      }
      __syncthreads();
      if (wave < 2) {          // attn @ V : K=128, one q-tile per wave
        int qt = wave;
        v8f acc = {};
#pragma unroll
        for (int kt = 0; kt < 4; ++kt)
          acc = wmma_f16(frag_a_f(&sc[qt*16][kt*32], 128, lane, 32),
                         frag_b_kn(&kv[1][kt*32][h*16], 64, lane, 32), acc);
#pragma unroll
        for (int r = 0; r < 8; ++r)
          os[(qblk*32 + qt*16 + hl*8 + r)*64 + h*16 + nloc] = (_Float16)acc[r];
      }
      __syncthreads();
    }
    __syncthreads();
  }
}

// ----------------------- output proj + pooling + agg ------------------------
__global__ void attn_out_kernel(const _Float16* __restrict__ oin,
                                const _Float16* __restrict__ wo, const float* __restrict__ bo,
                                const _Float16* __restrict__ f,
                                const float* __restrict__ a0w, const float* __restrict__ a0b,
                                const float* __restrict__ a1w, const float* __restrict__ a1b,
                                const int* __restrict__ cntL, const int* __restrict__ cntR,
                                float* __restrict__ lr) {
  __shared__ float ob[128][64];
  __shared__ float pre[64];
  __shared__ float r1[64];
  int slice = blockIdx.x;
  int wave = threadIdx.x >> 5, lane = threadIdx.x & 31;
  int nloc = lane & 15, hl = lane >> 4;
  int cnt = (slice < NB*NS) ? cntL[slice] : cntR[slice - NB*NS];
  const _Float16* os = oin + slice * NE * 64;

  v16h A0 = frag_a_h(os + wave*16*64,      64, lane, 32);
  v16h A1 = frag_a_h(os + wave*16*64 + 32, 64, lane, 32);
#pragma unroll
  for (int nt = 0; nt < 4; ++nt) {
    v8f acc = {};
    acc = wmma_f16(A0, frag_b_nk(wo + nt*16*64,      64, lane, 32), acc);
    acc = wmma_f16(A1, frag_b_nk(wo + nt*16*64 + 32, 64, lane, 32), acc);
    int n = nt*16 + nloc; float bias = bo[n];
#pragma unroll
    for (int r = 0; r < 8; ++r)
      ob[wave*16 + hl*8 + r][n] = acc[r] + bias;
  }
  __syncthreads();
  if (threadIdx.x < 64) {       // masked max pool over events
    int j = threadIdx.x;
    float mx = -1e9f;
    for (int e = 0; e < NE; ++e) if (e < cnt) mx = fmaxf(mx, ob[e][j]);
    float f0 = (float)f[slice * NE * 64 + j];   // embedded event 0
    pre[j] = (cnt == 1) ? f0 : mx;
  }
  __syncthreads();
  if (threadIdx.x < 64) {       // agg layer 1 (relu)
    int j = threadIdx.x; float s = a0b[j];
    for (int k = 0; k < 64; ++k) s += a0w[j*64 + k] * pre[k];
    r1[j] = fmaxf(s, 0.f);
  }
  __syncthreads();
  if (threadIdx.x < 64) {       // agg layer 2, zero empty slices
    int j = threadIdx.x; float s = a1b[j];
    for (int k = 0; k < 64; ++k) s += a1w[j*64 + k] * r1[k];
    lr[slice * 64 + j] = (cnt == 0) ? 0.f : s;
  }
}

// --------------------------------- fusion -----------------------------------
__global__ void fuse_kernel(const float* __restrict__ lr,
                            const float* __restrict__ fw,
                            float* __restrict__ fusedb) {
  int i = blockIdx.x * 256 + threadIdx.x;
  if (i >= NB*NS*64) return;
  int bs = i >> 6, j = i & 63;
  const float* lf = lr + bs*64;
  const float* rf = lr + NB*NS*64 + bs*64;
  float s = 0.f;
  for (int k = 0; k < 64; ++k)
    s += lf[k]*fw[j*128 + k] + rf[k]*fw[j*128 + 64 + k];
  fusedb[i] = s;
}

// ----------------------- sequential ODE scan over slices --------------------
// Single workgroup, 16 waves (512 threads): one WMMA n-tile per wave for the
// N=256 layers -> 4 independent wmma chains per SIMD32 to hide XDL latency on
// the serial critical path. M=16 (batch) = one WMMA M-tile. dt = 1.0.
__global__ void scan_kernel(const float* __restrict__ fusedb, const float* __restrict__ h0,
    const float* __restrict__ ot0w, const float* __restrict__ ot0b,
    const float* __restrict__ ot1w, const float* __restrict__ ot1b,
    const _Float16* __restrict__ wd0, const float* __restrict__ bd0,
    const _Float16* __restrict__ wd1, const float* __restrict__ bd1,
    const _Float16* __restrict__ wd2, const float* __restrict__ bd2,
    const _Float16* __restrict__ wu0, const float* __restrict__ bu0,
    const _Float16* __restrict__ wu1, const float* __restrict__ bu1,
    const _Float16* __restrict__ wg,  const float* __restrict__ bg,
    const _Float16* __restrict__ wt0, const float* __restrict__ bt0,
    const float* __restrict__ wt1,    const float* __restrict__ bt1,
    const _Float16* __restrict__ wr0, const float* __restrict__ br0,
    const float* __restrict__ wr1,    const float* __restrict__ br1,
    float* __restrict__ out) {
  __shared__ float    hf[16][256];
  __shared__ _Float16 a16[16][320];
  __shared__ _Float16 b16[16][256];
  __shared__ float    ttmp[32];
  __shared__ float    te[32];
  __shared__ float    pose[16][7];
  const int NT = 512;                      // block size
  int tid = threadIdx.x, wave = tid >> 5, lane = tid & 31;
  int nloc = lane & 15, hl = lane >> 4;

  for (int i = tid; i < 16*256; i += NT) hf[i >> 8][i & 255] = h0[i & 255];
  __syncthreads();

  for (int s = 0; s < NS; ++s) {
    // Warm L2->WGP path for next slice's fused event features while computing.
    if (s + 1 < NS && tid < 16)
      __builtin_prefetch(&fusedb[(tid*NS + s + 1)*64], 0, 0);

    // ---- 5 Euler steps: h += dt * ode_func(t, h), dt = 1 ----
    for (int st = 0; st < 5; ++st) {
      float t = s*5.0f + (float)st;
      if (tid < 32) ttmp[tid] = tanhf(ot0w[tid]*t + ot0b[tid]);
      __syncthreads();
      if (tid < 32) {
        float a = ot1b[tid];
        for (int j = 0; j < 32; ++j) a += ot1w[tid*32 + j] * ttmp[j];
        te[tid] = a;
      }
      __syncthreads();
      for (int i = tid; i < 16*288; i += NT) {
        int m = i / 288, k = i % 288;
        a16[m][k] = (k < 256) ? (_Float16)hf[m][k] : (_Float16)te[k - 256];
      }
      __syncthreads();
      { // d1 = tanh(Wd0 x + b), K=288; n-tile = wave; prefetch Wd1 panel
        int nt = wave;
        __builtin_prefetch(wd1 + nt*16*256, 0, 0);
        v8f acc = {};
#pragma unroll
        for (int kt = 0; kt < 9; ++kt)
          acc = wmma_f16(frag_a_h(&a16[0][kt*32], 320, lane, 32),
                         frag_b_nk(wd0 + nt*16*288 + kt*32, 288, lane, 32), acc);
        int n = nt*16 + nloc; float bias = bd0[n];
#pragma unroll
        for (int r = 0; r < 8; ++r) b16[hl*8 + r][n] = (_Float16)tanhf(acc[r] + bias);
      }
      __syncthreads();
      { // d2 = tanh(Wd1 d1), K=256
        int nt = wave;
        __builtin_prefetch(wd2 + nt*16*256, 0, 0);
        v8f acc = {};
#pragma unroll
        for (int kt = 0; kt < 8; ++kt)
          acc = wmma_f16(frag_a_h(&b16[0][kt*32], 256, lane, 32),
                         frag_b_nk(wd1 + nt*16*256 + kt*32, 256, lane, 32), acc);
        int n = nt*16 + nloc; float bias = bd1[n];
#pragma unroll
        for (int r = 0; r < 8; ++r) a16[hl*8 + r][n] = (_Float16)tanhf(acc[r] + bias);
      }
      __syncthreads();
      { // h += Wd2 d2 + b (dt=1)
        int nt = wave;
        v8f acc = {};
#pragma unroll
        for (int kt = 0; kt < 8; ++kt)
          acc = wmma_f16(frag_a_h(&a16[0][kt*32], 320, lane, 32),
                         frag_b_nk(wd2 + nt*16*256 + kt*32, 256, lane, 32), acc);
        int n = nt*16 + nloc; float bias = bd2[n];
#pragma unroll
        for (int r = 0; r < 8; ++r) hf[hl*8 + r][n] += acc[r] + bias;
      }
      __syncthreads();
    }
    // ---- gated event update ----
    for (int i = tid; i < 16*320; i += NT) {
      int m = i / 320, k = i % 320;
      a16[m][k] = (k < 256) ? (_Float16)hf[m][k]
                            : (_Float16)fusedb[(m*NS + s)*64 + (k - 256)];
    }
    __syncthreads();
    { // u1 = relu(Wu0 [h,ef] + b), K=320
      int nt = wave;
      __builtin_prefetch(wu1 + nt*16*256, 0, 0);
      v8f acc = {};
#pragma unroll
      for (int kt = 0; kt < 10; ++kt)
        acc = wmma_f16(frag_a_h(&a16[0][kt*32], 320, lane, 32),
                       frag_b_nk(wu0 + nt*16*320 + kt*32, 320, lane, 32), acc);
      int n = nt*16 + nloc; float bias = bu0[n];
#pragma unroll
      for (int r = 0; r < 8; ++r) b16[hl*8 + r][n] = (_Float16)fmaxf(acc[r] + bias, 0.f);
    }
    __syncthreads();
    { // h += sigmoid(Wg ef + b) * (Wu1 u1 + b)
      int nt = wave;
      v8f au = {}; v8f ag = {};
#pragma unroll
      for (int kt = 0; kt < 8; ++kt)
        au = wmma_f16(frag_a_h(&b16[0][kt*32], 256, lane, 32),
                      frag_b_nk(wu1 + nt*16*256 + kt*32, 256, lane, 32), au);
#pragma unroll
      for (int kt = 0; kt < 2; ++kt)
        ag = wmma_f16(frag_a_h(&a16[0][256 + kt*32], 320, lane, 32),
                      frag_b_nk(wg + nt*16*64 + kt*32, 64, lane, 32), ag);
      int n = nt*16 + nloc;
      float bu = bu1[n], bgn = bg[n];
#pragma unroll
      for (int r = 0; r < 8; ++r) {
        float gate = 1.f / (1.f + __expf(-(ag[r] + bgn)));
        hf[hl*8 + r][n] += gate * (au[r] + bu);
      }
    }
    __syncthreads();
    for (int i = tid; i < 16*256; i += NT) {
      int m = i >> 8, k = i & 255;
      a16[m][k] = (_Float16)hf[m][k];
    }
    __syncthreads();
    { // trans (waves 0-7) / rot (waves 8-15) hidden layers, N=128 each
      if (wave < 8) {
        int nt = wave;
        v8f at = {};
#pragma unroll
        for (int kt = 0; kt < 8; ++kt)
          at = wmma_f16(frag_a_h(&a16[0][kt*32], 320, lane, 32),
                        frag_b_nk(wt0 + nt*16*256 + kt*32, 256, lane, 32), at);
        int n = nt*16 + nloc; float b_t = bt0[n];
#pragma unroll
        for (int r = 0; r < 8; ++r)
          b16[hl*8 + r][n] = (_Float16)fmaxf(at[r] + b_t, 0.f);
      } else {
        int nt = wave - 8;
        v8f ar = {};
#pragma unroll
        for (int kt = 0; kt < 8; ++kt)
          ar = wmma_f16(frag_a_h(&a16[0][kt*32], 320, lane, 32),
                        frag_b_nk(wr0 + nt*16*256 + kt*32, 256, lane, 32), ar);
        int n = nt*16 + nloc; float b_r = br0[n];
#pragma unroll
        for (int r = 0; r < 8; ++r)
          b16[hl*8 + r][128 + n] = (_Float16)fmaxf(ar[r] + b_r, 0.f);
      }
    }
    __syncthreads();
    if (tid < 16*7) {    // pose heads (3 + 4 outputs)
      int m = tid / 7, j = tid % 7;
      float v;
      if (j < 3) {
        v = bt1[j];
        for (int k = 0; k < 128; ++k) v += wt1[j*128 + k] * (float)b16[m][k];
      } else {
        int jj = j - 3;
        v = br1[jj];
        for (int k = 0; k < 128; ++k) v += wr1[jj*128 + k] * (float)b16[m][128 + k];
      }
      pose[m][j] = v;
    }
    __syncthreads();
    if (tid < 16) {      // quaternion normalization + pose writeback
      int m = tid;
      float nrm = sqrtf(pose[m][3]*pose[m][3] + pose[m][4]*pose[m][4] +
                        pose[m][5]*pose[m][5] + pose[m][6]*pose[m][6]);
      float inv = 1.f / fmaxf(nrm, 1e-12f);
      int base = (m*NS + s) * 7;
      out[base+0] = pose[m][0];
      out[base+1] = pose[m][1];
      out[base+2] = pose[m][2];
      out[base+3] = pose[m][3]*inv;
      out[base+4] = pose[m][4]*inv;
      out[base+5] = pose[m][5]*inv;
      out[base+6] = pose[m][6]*inv;
    }
    for (int i = tid; i < 16*256; i += NT) {
      int m = i >> 8, k = i & 255;
      out[7*NB*NS + (m*NS + s)*256 + k] = hf[m][k];
    }
    __syncthreads();
  }
}

// ------------------------------- host launcher -------------------------------
extern "C" void kernel_launch(void* const* d_in, const int* in_sizes, int n_in,
                              void* d_out, int out_size, void* d_ws, size_t ws_size,
                              hipStream_t stream) {
  (void)in_sizes; (void)n_in; (void)out_size; (void)ws_size;
  const float* evL   = (const float*)d_in[0];
  const float* evR   = (const float*)d_in[1];
  const int*   cntL  = (const int*)d_in[2];
  const int*   cntR  = (const int*)d_in[3];
  const float* agg0_b=(const float*)d_in[4];  const float* agg0_w=(const float*)d_in[5];
  const float* agg1_b=(const float*)d_in[6];  const float* agg1_w=(const float*)d_in[7];
  const float* atK_b =(const float*)d_in[8];  const float* atK_w =(const float*)d_in[9];
  const float* atO_b =(const float*)d_in[10]; const float* atO_w =(const float*)d_in[11];
  const float* atQ_b =(const float*)d_in[12]; const float* atQ_w =(const float*)d_in[13];
  const float* atV_b =(const float*)d_in[14]; const float* atV_w =(const float*)d_in[15];
  const float* e0_b  =(const float*)d_in[16]; const float* e0_w  =(const float*)d_in[17];
  const float* e1_b  =(const float*)d_in[18]; const float* e1_w  =(const float*)d_in[19];
  const float* e2_b  =(const float*)d_in[20]; const float* e2_w  =(const float*)d_in[21];
  const float* g_b   =(const float*)d_in[22]; const float* g_w   =(const float*)d_in[23];
  const float* u0_b  =(const float*)d_in[24]; const float* u0_w  =(const float*)d_in[25];
  const float* u1_b  =(const float*)d_in[26]; const float* u1_w  =(const float*)d_in[27];
  const float* fuse_w=(const float*)d_in[28];
  const float* h0    =(const float*)d_in[29];
  const float* d0_b  =(const float*)d_in[30]; const float* d0_w  =(const float*)d_in[31];
  const float* d1_b  =(const float*)d_in[32]; const float* d1_w  =(const float*)d_in[33];
  const float* d2_b  =(const float*)d_in[34]; const float* d2_w  =(const float*)d_in[35];
  const float* ot0_b =(const float*)d_in[36]; const float* ot0_w =(const float*)d_in[37];
  const float* ot1_b =(const float*)d_in[38]; const float* ot1_w =(const float*)d_in[39];
  const float* r0_b  =(const float*)d_in[40]; const float* r0_w  =(const float*)d_in[41];
  const float* r1_b  =(const float*)d_in[42]; const float* r1_w  =(const float*)d_in[43];
  const float* t0_b  =(const float*)d_in[44]; const float* t0_w  =(const float*)d_in[45];
  const float* t1_b  =(const float*)d_in[46]; const float* t1_w  =(const float*)d_in[47];

  char* ws = (char*)d_ws;
  size_t off = 0;
  auto alloc = [&](size_t bytes) -> void* {
    off = (off + 255) & ~(size_t)255;
    void* p = ws + off; off += bytes; return p;
  };
  _Float16* w_e0 = (_Float16*)alloc((size_t)128*32*2);
  _Float16* w_e1 = (_Float16*)alloc((size_t)128*128*2);
  _Float16* w_e2 = (_Float16*)alloc((size_t)64*128*2);
  _Float16* w_q  = (_Float16*)alloc((size_t)64*64*2);
  _Float16* w_k  = (_Float16*)alloc((size_t)64*64*2);
  _Float16* w_v  = (_Float16*)alloc((size_t)64*64*2);
  _Float16* w_o  = (_Float16*)alloc((size_t)64*64*2);
  _Float16* w_g  = (_Float16*)alloc((size_t)256*64*2);
  _Float16* w_u0 = (_Float16*)alloc((size_t)256*320*2);
  _Float16* w_u1 = (_Float16*)alloc((size_t)256*256*2);
  _Float16* w_d0 = (_Float16*)alloc((size_t)256*288*2);
  _Float16* w_d1 = (_Float16*)alloc((size_t)256*256*2);
  _Float16* w_d2 = (_Float16*)alloc((size_t)256*256*2);
  _Float16* w_t0 = (_Float16*)alloc((size_t)128*256*2);
  _Float16* w_r0 = (_Float16*)alloc((size_t)128*256*2);
  _Float16* f16buf = (_Float16*)alloc((size_t)2*NB*NS*NE*64*2);
  _Float16* o16buf = (_Float16*)alloc((size_t)2*NB*NS*NE*64*2);
  float* lr     = (float*)alloc((size_t)2*NB*NS*64*4);
  float* fusedb = (float*)alloc((size_t)NB*NS*64*4);

  auto cvt = [&](const float* src, _Float16* dst, int rows, int cols, int pcols) {
    int total = rows * pcols;
    cvt_pad_kernel<<<(total + 255)/256, 256, 0, stream>>>(src, dst, rows, cols, pcols);
  };
  cvt(e0_w, w_e0, 128, 4, 32);
  cvt(e1_w, w_e1, 128, 128, 128);
  cvt(e2_w, w_e2, 64, 128, 128);
  cvt(atQ_w, w_q, 64, 64, 64);
  cvt(atK_w, w_k, 64, 64, 64);
  cvt(atV_w, w_v, 64, 64, 64);
  cvt(atO_w, w_o, 64, 64, 64);
  cvt(g_w,  w_g, 256, 64, 64);
  cvt(u0_w, w_u0, 256, 320, 320);
  cvt(u1_w, w_u1, 256, 256, 256);
  cvt(d0_w, w_d0, 256, 288, 288);
  cvt(d1_w, w_d1, 256, 256, 256);
  cvt(d2_w, w_d2, 256, 256, 256);
  cvt(t0_w, w_t0, 128, 256, 256);
  cvt(r0_w, w_r0, 128, 256, 256);

  embed_kernel<<<(2*NB*NS*NE/16)/4, 128, 0, stream>>>(
      evL, evR, w_e0, e0_b, w_e1, e1_b, w_e2, e2_b, f16buf);
  attn_kernel<<<2*NB*NS, 256, 0, stream>>>(
      f16buf, w_q, atQ_b, w_k, atK_b, w_v, atV_b, cntL, cntR, o16buf);
  attn_out_kernel<<<2*NB*NS, 256, 0, stream>>>(
      o16buf, w_o, atO_b, f16buf, agg0_w, agg0_b, agg1_w, agg1_b, cntL, cntR, lr);
  fuse_kernel<<<(NB*NS*64 + 255)/256, 256, 0, stream>>>(lr, fuse_w, fusedb);
  scan_kernel<<<1, 512, 0, stream>>>(
      fusedb, h0, ot0_w, ot0_b, ot1_w, ot1_b,
      w_d0, d0_b, w_d1, d1_b, w_d2, d2_b,
      w_u0, u0_b, w_u1, u1_b, w_g, g_b,
      w_t0, t0_b, t1_w, t1_b, w_r0, r0_b, r1_w, r1_b,
      (float*)d_out);
}
